// RGATLayer_62861141344356
// MI455X (gfx1250) — matrix operand order repaired and checked by reading, not compiled, not run on hardware
//
#include <hip/hip_runtime.h>

#define NN    3072
#define DIN   128
#define DOUT  128
#define RR    3
#define HH    4
#define HD    32
#define UCOLS 144   // 128 u-cols + 4 e-cols + 1 ones-col, padded to 9*16
#define KTILE 32
#define NTILES (NN / KTILE)   // 96

typedef __attribute__((ext_vector_type(16))) _Float16 v16h;
typedef __attribute__((ext_vector_type(8)))  _Float16 v8h;
typedef __attribute__((ext_vector_type(8)))  float    v8f;
typedef __attribute__((ext_vector_type(4)))  int      v4i;

#if defined(__HIP_DEVICE_COMPILE__) && \
    __has_builtin(__builtin_amdgcn_global_load_async_to_lds_b128) && \
    __has_builtin(__builtin_amdgcn_s_wait_asynccnt)
#define USE_ASYNC_LDS 1
#else
#define USE_ASYNC_LDS 0
#endif

#define AS1 __attribute__((address_space(1)))
#define AS3 __attribute__((address_space(3)))
typedef AS1 v4i* gv4i_p;   // global-AS pointer to int4 (64-bit)
typedef AS3 v4i* lv4i_p;   // LDS-AS pointer to int4 (32-bit)

// ---------------- Kernel 0: convert X and W to f16 ----------------
__global__ void k0_convert(const float* __restrict__ X, const float* __restrict__ W,
                           _Float16* __restrict__ Xh, _Float16* __restrict__ Wh) {
    int i = blockIdx.x * blockDim.x + threadIdx.x;
    const int nX = NN * DIN;
    const int nW = RR * DOUT * DIN;
    if (i < nX)            Xh[i]      = (_Float16)X[i];
    else if (i < nX + nW)  Wh[i - nX] = (_Float16)W[i - nX];
}

// ---------------- Kernel 1: t[r] = X * W_r^T via WMMA f16 ----------------
// One wave per 16x16 output tile; K=128 in 4 wmma steps.
__global__ void __launch_bounds__(128) k1_transform(const _Float16* __restrict__ Xh,
                                                    const _Float16* __restrict__ Wh,
                                                    float* __restrict__ t) {
    int lane = threadIdx.x & 31;
    int wave = threadIdx.x >> 5;
    int tile = blockIdx.x * 4 + wave;
    const int TO = DOUT / 16;            // 8
    const int TM = NN / 16;              // 192
    int r   = tile / (TM * TO);
    int rem = tile % (TM * TO);
    int m0  = (rem / TO) * 16;
    int o0  = (rem % TO) * 16;
    int hi  = lane >> 4;                 // lane half: selects K sub-ranges
    int l15 = lane & 15;

    const _Float16* xrow = Xh + (size_t)(m0 + l15) * DIN;
    const _Float16* wrow = Wh + (size_t)(r * DOUT + o0 + l15) * DIN;

    v8f c = {};
#pragma unroll
    for (int k0 = 0; k0 < DIN; k0 += 32) {
        // A fragment (16x32 f16): lane<16 holds K {0..15}, lane>=16 holds {16..31}
        v8h a_lo = *(const v8h*)(xrow + k0 + hi * 16);
        v8h a_hi = *(const v8h*)(xrow + k0 + hi * 16 + 8);
        v16h a;
#pragma unroll
        for (int q = 0; q < 8; ++q) { a[q] = a_lo[q]; a[q + 8] = a_hi[q]; }
        // B fragment (32x16): B[k,n] = W[o0+n, k] -> contiguous 16 halves along K
        v16h b = *(const v16h*)(wrow + k0 + hi * 16);
        c = __builtin_amdgcn_wmma_f32_16x16x32_f16(false, a, false, b, (short)0, c, false, false);
    }
    int crow = m0 + hi * 8;
    int ccol = o0 + l15;
#pragma unroll
    for (int v = 0; v < 8; ++v)
        t[(size_t)(r * NN + crow + v) * DOUT + ccol] = c[v];
}

// ---------------- Kernel 2: attention scalars + build transposed U ----------------
// Ut[r][col][n]: col 0..127 -> exp(s_dst_h)*t, 128..131 -> exp(s_dst_h), 132 -> 1, rest 0.
__global__ void k2_attn(const float* __restrict__ t, const float* __restrict__ avec,
                        _Float16* __restrict__ Ut, float* __restrict__ S) {
    int idx = blockIdx.x * blockDim.x + threadIdx.x;
    if (idx >= RR * NN) return;
    int r = idx / NN, n = idx % NN;
    const float* trow = t + (size_t)(r * NN + n) * DOUT;
    const float* ar   = avec + r * (2 * HD);

    float e[HH];
#pragma unroll
    for (int h = 0; h < HH; ++h) {
        float ss = 0.f, sd = 0.f;
        for (int d = 0; d < HD; ++d) {
            float tv = trow[h * HD + d];
            ss += tv * ar[d];
            sd += tv * ar[HD + d];
        }
        S[(size_t)(r * NN + n) * HH + h] = __expf(ss);   // exp(s_src)
        e[h] = __expf(sd);                               // exp(s_dst)
    }
    _Float16* ub = Ut + (size_t)r * UCOLS * NN;
    for (int col = 0; col < DOUT; ++col)
        ub[(size_t)col * NN + n] = (_Float16)(e[col / HD] * trow[col]);
#pragma unroll
    for (int h = 0; h < HH; ++h)
        ub[(size_t)(DOUT + h) * NN + n] = (_Float16)e[h];
    ub[(size_t)(DOUT + HH) * NN + n] = (_Float16)1.0f;   // ones column -> degree
    for (int col = DOUT + HH + 1; col < UCOLS; ++col)
        ub[(size_t)col * NN + n] = (_Float16)0.0f;
}

// ---------------- Kernel 3: agg[r] = A_r (0/1) @ U_r via WMMA ----------------
// Block: 288 threads = 9 waves; block owns 16 output rows x all 144 cols.
// Raw int32 adjacency tiles (16x32 = 2KB) stream through a double-buffered LDS
// stage; on CDNA5 the copy is GLOBAL_LOAD_ASYNC_TO_LDS_B128 issued by wave 0
// (one tile kept in flight, s_wait_asynccnt 4 before consuming the current one).
// 0/1 ints are converted to f16 in registers while building the A fragment.
__global__ void __launch_bounds__(288) k3_agg(const int* __restrict__ adj,
                                              const _Float16* __restrict__ Ut,
                                              float* __restrict__ agg) {
    __shared__ __align__(16) int sAi[2][16 * KTILE];   // 2 x 2KB raw adjacency tiles
    int tid  = threadIdx.x;
    int lane = tid & 31;
    int wave = tid >> 5;                 // 0..8 -> column tile
    int r  = blockIdx.x / (NN / 16);
    int i0 = (blockIdx.x % (NN / 16)) * 16;
    int hi  = lane >> 4;
    int l15 = lane & 15;
    int c0  = wave * 16;

    const _Float16* ubase = Ut + ((size_t)r * UCOLS + c0 + l15) * NN;
    const int* abase = adj + (size_t)(r * NN + i0) * NN;

    // Stage tile `it` (K range [it*32, it*32+32)) into buffer `buf`.
    auto stage = [&](int it, int buf) {
#if USE_ASYNC_LDS
        if (wave == 0) {
#pragma unroll
            for (int q = 0; q < 4; ++q) {
                int chunk = q * 32 + lane;            // 128 chunks of 16B
                int row   = chunk >> 3;
                int ci    = (chunk & 7) * 4;          // int index within row
                const int* gp = abase + (size_t)row * NN + it * KTILE + ci;
                void*      lp = (void*)&sAi[buf][chunk * 4];
                __builtin_amdgcn_global_load_async_to_lds_b128(
                    (gv4i_p)(uintptr_t)gp,
                    (lv4i_p)(uint32_t)(uintptr_t)lp, 0, 0);
            }
        }
#else
        if (tid < 128) {
            int chunk = tid;
            int row   = chunk >> 3;
            int ci    = (chunk & 7) * 4;
            *(v4i*)&sAi[buf][chunk * 4] =
                *(const v4i*)(abase + (size_t)row * NN + it * KTILE + ci);
        }
#endif
    };

    stage(0, 0);
    v8f c = {};
    for (int it = 0; it < NTILES; ++it) {
        int cur = it & 1;
        if (it + 1 < NTILES) stage(it + 1, cur ^ 1);
#if USE_ASYNC_LDS
        if (wave == 0) {
            if (it + 1 < NTILES) __builtin_amdgcn_s_wait_asynccnt(4); // tile `it` landed
            else                 __builtin_amdgcn_s_wait_asynccnt(0);
        }
#endif
        __syncthreads();
        // A fragment straight from the raw int tile (convert 0/1 -> f16 in regs)
        const v4i* pa = (const v4i*)&sAi[cur][l15 * KTILE + hi * 16];
        v4i w0 = pa[0], w1 = pa[1];                    // K {0..7} or {16..23}
        const v4i* pb = (const v4i*)&sAi[cur][l15 * KTILE + hi * 16 + 8];
        v4i w2 = pb[0], w3 = pb[1];                    // K {8..15} or {24..31}
        v16h a;
#pragma unroll
        for (int q = 0; q < 4; ++q) {
            a[q]      = (_Float16)(w0[q] != 0 ? 1.0f : 0.0f);
            a[q + 4]  = (_Float16)(w1[q] != 0 ? 1.0f : 0.0f);
            a[q + 8]  = (_Float16)(w2[q] != 0 ? 1.0f : 0.0f);
            a[q + 12] = (_Float16)(w3[q] != 0 ? 1.0f : 0.0f);
        }
        v16h b = *(const v16h*)(ubase + it * KTILE + hi * 16);   // contiguous 32B
        c = __builtin_amdgcn_wmma_f32_16x16x32_f16(false, a, false, b, (short)0, c, false, false);
        __syncthreads();
    }
    int crow = i0 + hi * 8;
    int ccol = c0 + l15;
#pragma unroll
    for (int v = 0; v < 8; ++v)
        agg[(size_t)(r * NN + crow + v) * UCOLS + ccol] = c[v];
}

// ---------------- Kernel 4: normalize by Z, mean over relations, + bias ----------------
__global__ void k4_finalize(const float* __restrict__ agg, const float* __restrict__ S,
                            const float* __restrict__ bias, float* __restrict__ out) {
    int idx = blockIdx.x * blockDim.x + threadIdx.x;
    if (idx >= NN * DOUT) return;
    int i = idx / DOUT, o = idx % DOUT;
    int h = o / HD;
    float acc = 0.f;
#pragma unroll
    for (int r = 0; r < RR; ++r) {
        const float* ag = agg + (size_t)(r * NN + i) * UCOLS;
        float esrc = S[(size_t)(r * NN + i) * HH + h];
        float E    = ag[DOUT + h];         // sum of exp(s_dst) over edges
        float deg  = ag[DOUT + HH];        // degree (exact in f32)
        float Z    = esrc * E + ((float)NN - deg);  // non-edges contribute exp(0)=1
        acc += (esrc / Z) * ag[o];
    }
    out[idx] = acc * (1.0f / (float)RR) + bias[o];
}

extern "C" void kernel_launch(void* const* d_in, const int* in_sizes, int n_in,
                              void* d_out, int out_size, void* d_ws, size_t ws_size,
                              hipStream_t stream) {
    const float* X    = (const float*)d_in[0];
    const int*   adj  = (const int*)  d_in[1];
    const float* W    = (const float*)d_in[2];
    const float* avec = (const float*)d_in[3];
    const float* bias = (const float*)d_in[4];
    float* out = (float*)d_out;

    char* ws = (char*)d_ws;
    size_t off = 0;
    auto alloc = [&](size_t bytes) -> char* {
        char* p = ws + off;
        off = (off + bytes + 255) & ~(size_t)255;
        return p;
    };
    _Float16* Xh = (_Float16*)alloc(sizeof(_Float16) * NN * DIN);
    _Float16* Wh = (_Float16*)alloc(sizeof(_Float16) * RR * DOUT * DIN);
    float*    t  = (float*)   alloc(sizeof(float)    * RR * NN * DOUT);
    _Float16* Ut = (_Float16*)alloc(sizeof(_Float16) * RR * UCOLS * NN);
    float*    S  = (float*)   alloc(sizeof(float)    * RR * NN * HH);
    float*    agg= (float*)   alloc(sizeof(float)    * RR * NN * UCOLS);

    int nconv = NN * DIN + RR * DOUT * DIN;
    k0_convert  <<<(nconv + 255) / 256, 256, 0, stream>>>(X, W, Xh, Wh);
    k1_transform<<<(RR * (NN / 16) * (DOUT / 16)) / 4, 128, 0, stream>>>(Xh, Wh, t);
    k2_attn     <<<(RR * NN + 127) / 128, 128, 0, stream>>>(t, avec, Ut, S);
    k3_agg      <<<RR * (NN / 16), 288, 0, stream>>>(adj, Ut, agg);
    k4_finalize <<<(NN * DOUT + 255) / 256, 256, 0, stream>>>(agg, S, bias, out);
}
